// CoAttention_57157424775723
// MI455X (gfx1250) — matrix operand order repaired and checked by reading, not compiled, not run on hardware
//
#include <hip/hip_runtime.h>
#include <math.h>

// ---------------------------------------------------------------------------
// CoAttention GRU for MI455X (gfx1250): bf16 WMMA GEMMs + async-LDS gather
// double buffering + fused co-attention head.
// ---------------------------------------------------------------------------

#define VOCAB 50000
#define E     300
#define H     300
#define L     75
#define BATCH 2048
#define OUT   5

#define BT   16      // batch tile (WMMA M)
#define KP   320     // K padded (300 -> 320, multiple of 32)
#define NP   304     // per-gate N padded (300 -> 304, multiple of 16)
#define XSP  304     // x staging row stride (floats), 1216 B, 16B aligned
#define GSTR (3*NP)  // 912 padded gate row
#define NTILES (GSTR/16)  // 57

typedef __attribute__((ext_vector_type(16))) __bf16 v16bf;
typedef __attribute__((ext_vector_type(8)))  __bf16 bf16x8;
typedef __attribute__((ext_vector_type(8)))  float  v8f;

union FragA { v16bf v; bf16x8 h[2]; };

__device__ __forceinline__ __bf16 f32_to_bf16(float f) {
  unsigned u = __builtin_bit_cast(unsigned, f);
  u = (u + 0x7FFFu + ((u >> 16) & 1u)) >> 16;       // round-to-nearest-even
  unsigned short s = (unsigned short)u;
  return __builtin_bit_cast(__bf16, s);
}

__device__ __forceinline__ float fast_sigmoid(float x) {
  return 1.0f / (1.0f + __expf(-x));
}
__device__ __forceinline__ float fast_tanh(float x) {
  float e = __expf(-2.0f * x);
  return (1.0f - e) / (1.0f + e);
}

// LDS byte offset of a pointer into a __shared__ object (addrspace(3) cast).
__device__ __forceinline__ unsigned lds_off(const void* p) {
  return (unsigned)(unsigned long long)(__attribute__((address_space(3))) const void*)p;
}

// One 16-byte async gather lane: LDS[loff] = *gptr (tracked by ASYNCcnt).
__device__ __forceinline__ void async_copy_b128(unsigned loff, const void* gptr) {
  asm volatile("global_load_async_to_lds_b128 %0, %1, off"
               :: "v"(loff), "v"(gptr)
               : "memory");
}
__device__ __forceinline__ void wait_async0() {
  asm volatile("s_wait_asynccnt 0x0" ::: "memory");
}

// ---------------------------------------------------------------------------
// Kernel 0: convert W (900x300 f32, gate-major rows) into padded bf16
// Wp[3][NP][KP]; rows 300..303 and cols 300..319 of each gate are zero.
// ---------------------------------------------------------------------------
__global__ void prep_w_kernel(const float* __restrict__ W,
                              unsigned short* __restrict__ Wp) {
  int i = blockIdx.x * blockDim.x + threadIdx.x;
  if (i >= 3 * NP * KP) return;
  int k = i % KP;
  int n = (i / KP) % NP;
  int g = i / (KP * NP);
  float v = 0.0f;
  if (n < H && k < E) v = W[(g * H + n) * E + k];
  Wp[i] = __builtin_bit_cast(unsigned short, f32_to_bf16(v));
}

// ---------------------------------------------------------------------------
// Kernel 1: fused embedding gather + GRU over 75 steps. One WG per 16 batch
// rows; 8 waves split the 57 N-tiles; two bf16 WMMA GEMM streams (x@Wih^T,
// h@Whh^T) with f32 accumulation; next step's embedding rows stream into LDS
// asynchronously (double buffered) while the current step computes.
// ---------------------------------------------------------------------------
__global__ __launch_bounds__(256) void gru_kernel(
    const int*   __restrict__ seq,     // (B, L)
    const float* __restrict__ emb,     // (VOCAB, E)
    const unsigned short* __restrict__ WihP,  // [3][NP][KP] bf16
    const unsigned short* __restrict__ WhhP,  // [3][NP][KP] bf16
    const float* __restrict__ b_ih,    // (3H)
    const float* __restrict__ b_hh,    // (3H)
    float*       __restrict__ h_out)   // (B, L, H)
{
  __shared__ __align__(16) __bf16 xbf[BT][KP];
  __shared__ __align__(16) __bf16 hbf[BT][KP];
  __shared__ __align__(16) float  hfp[BT][H];
  __shared__ __align__(16) float  gx[BT][GSTR];
  __shared__ __align__(16) float  gh[BT][GSTR];
  __shared__ __align__(16) float  xstage[2][BT][XSP];   // async fp32 staging

  const int tid   = threadIdx.x;
  const int lane  = tid & 31;
  const int wave  = tid >> 5;
  const int bBase = blockIdx.x * BT;

  // zero hidden state + K-pad regions (pads stay zero for the whole kernel)
  for (int i = tid; i < BT * KP; i += 256) {
    (&hbf[0][0])[i] = f32_to_bf16(0.0f);
    (&xbf[0][0])[i] = f32_to_bf16(0.0f);
  }
  for (int i = tid; i < BT * H; i += 256) (&hfp[0][0])[i] = 0.0f;

  // ---- prologue: async-gather x_0 into stage 0 ----
  for (int i = tid; i < BT * (E / 4); i += 256) {       // 16 rows x 75 chunks
    int m = i / (E / 4), c4 = i % (E / 4);
    int tok = seq[(bBase + m) * L + 0];
    async_copy_b128(lds_off(&xstage[0][m][c4 * 4]),
                    emb + (size_t)tok * E + c4 * 4);
  }
  wait_async0();
  __syncthreads();
  for (int i = tid; i < BT * E; i += 256) {
    int m = i / E, e = i % E;
    xbf[m][e] = f32_to_bf16(xstage[0][m][e]);
  }
  __syncthreads();

  // A-fragment lane addressing (16-bit A 16x32 layout)
  const int mRow   = lane & 15;
  const int kHalfA = (lane >> 4) * 8;    // A: chunks at kh and kh+16
  const int kHalfB = (lane >> 4) * 16;   // unused; B loads use kHalfA chunks like A
  const __bf16* wih = (const __bf16*)WihP;
  const __bf16* whh = (const __bf16*)WhhP;

  for (int t = 0; t < L; ++t) {
    // ---- 0. kick off async gather of x_{t+1} into the other stage buffer ----
    const int nb = (t + 1) & 1;
    if (t + 1 < L) {
      for (int i = tid; i < BT * (E / 4); i += 256) {
        int m = i / (E / 4), c4 = i % (E / 4);
        int tok = seq[(bBase + m) * L + (t + 1)];
        async_copy_b128(lds_off(&xstage[nb][m][c4 * 4]),
                        emb + (size_t)tok * E + c4 * 4);
      }
    }

    // ---- 1. gate GEMMs: gx = x@Wih^T, gh = h@Whh^T (bf16 WMMA, f32 acc) ----
    for (int nt = wave; nt < NTILES; nt += 8) {
      const int g  = nt / (NP / 16);
      const int ti = nt % (NP / 16);
      const int nRow = g * NP + ti * 16 + mRow;     // row of padded W
      const __bf16* wi = wih + (size_t)nRow * KP;
      const __bf16* wh = whh + (size_t)nRow * KP;

      v8f accx = {};
      v8f acch = {};
      #pragma unroll
      for (int kb = 0; kb < KP / 32; ++kb) {
        const int ka  = kb * 32 + kHalfA;
        const int kbb = kb * 32 + kHalfA;
        FragA ax, ah, bi, bh;
        ax.h[0] = *(const bf16x8*)&xbf[mRow][ka];
        ax.h[1] = *(const bf16x8*)&xbf[mRow][ka + 16];
        ah.h[0] = *(const bf16x8*)&hbf[mRow][ka];
        ah.h[1] = *(const bf16x8*)&hbf[mRow][ka + 16];
        bi.h[0] = *(const bf16x8*)(wi + kbb);
        bi.h[1] = *(const bf16x8*)(wi + kbb + 16);
        bh.h[0] = *(const bf16x8*)(wh + kbb);
        bh.h[1] = *(const bf16x8*)(wh + kbb + 16);
        accx = __builtin_amdgcn_wmma_f32_16x16x32_bf16(
            false, ax.v, false, bi.v, (short)0, accx, false, false);
        acch = __builtin_amdgcn_wmma_f32_16x16x32_bf16(
            false, ah.v, false, bh.v, (short)0, acch, false, false);
      }
      // D layout: vgpr i -> M = i + 8*(lane>=16), N = lane&15
      const int nCol  = g * NP + ti * 16 + (lane & 15);
      const int mBase = (lane >> 4) * 8;
      #pragma unroll
      for (int i2 = 0; i2 < 8; ++i2) {
        gx[mBase + i2][nCol] = accx[i2];
        gh[mBase + i2][nCol] = acch[i2];
      }
    }
    __syncthreads();

    // ---- 2. pointwise GRU update ----
    for (int i = tid; i < BT * H; i += 256) {
      int m = i / H, c = i % H;
      float xr = gx[m][c]          + b_ih[c];
      float xz = gx[m][NP + c]     + b_ih[H + c];
      float xn = gx[m][2 * NP + c] + b_ih[2 * H + c];
      float hr = gh[m][c]          + b_hh[c];
      float hz = gh[m][NP + c]     + b_hh[H + c];
      float hn = gh[m][2 * NP + c] + b_hh[2 * H + c];
      float r  = fast_sigmoid(xr + hr);
      float z  = fast_sigmoid(xz + hz);
      float nn = fast_tanh(xn + r * hn);
      float hnew = (1.0f - z) * nn + z * hfp[m][c];
      hfp[m][c] = hnew;
      hbf[m][c] = f32_to_bf16(hnew);
      h_out[((size_t)(bBase + m) * L + t) * H + c] = hnew;
    }

    // ---- 3. land async gather, convert stage -> xbf for next step ----
    if (t + 1 < L) {
      wait_async0();            // this wave's async copies are complete
      __syncthreads();          // everyone's copies + pointwise complete
      for (int i = tid; i < BT * E; i += 256) {
        int m = i / E, e = i % E;
        xbf[m][e] = f32_to_bf16(xstage[nb][m][e]);
      }
    }
    __syncthreads();
  }
}

// ---------------------------------------------------------------------------
// Kernel 2: co-attention head, fully fused per batch element. Uses the
// factorization sm_i = S1*a_i + S2 + (h_i*wc)·u + cb*S1
//                      - Wmy_i*(a_i + b_i + c_ii + cb) + Wmy_b
// so the (L,L) score matrix is never materialized.
// ---------------------------------------------------------------------------
__global__ __launch_bounds__(256) void coattn_kernel(
    const float* __restrict__ h_all,    // (B, L, H)
    const float* __restrict__ wCo_w,    // (3H)
    const float* __restrict__ wCo_b,    // (1)
    const float* __restrict__ Wmy_w,    // (L)
    const float* __restrict__ Wmy_b,    // (1)
    const float* __restrict__ logits_w, // (OUT, H)
    const float* __restrict__ logits_b, // (OUT)
    float*       __restrict__ out)      // (B, OUT)
{
  __shared__ float hs[L][H];
  __shared__ float av[L], bv[L], qv[L], pv[L], smv[L], attn[L];
  __shared__ float u[H], pooled[H];
  __shared__ float red[2];

  const int b = blockIdx.x;
  const int tid = threadIdx.x;
  const float* hb = h_all + (size_t)b * L * H;
  for (int i = tid; i < L * H; i += 256) (&hs[0][0])[i] = hb[i];
  __syncthreads();

  const float* wa  = wCo_w;
  const float* wbp = wCo_w + H;
  const float* wc  = wCo_w + 2 * H;

  for (int d = tid; d < H; d += 256) {        // u = h^T @ Wmy
    float s = 0.0f;
    for (int i = 0; i < L; ++i) s += Wmy_w[i] * hs[i][d];
    u[d] = s;
  }
  if (tid < L) {                              // per-row reductions
    float sa = 0.0f, sb = 0.0f, sq = 0.0f;
    for (int d = 0; d < H; ++d) {
      float hv = hs[tid][d];
      sa += hv * wa[d];
      sb += hv * wbp[d];
      sq += hv * hv * wc[d];
    }
    av[tid] = sa; bv[tid] = sb; qv[tid] = sq;
  }
  __syncthreads();
  if (tid < L) {
    float sp = 0.0f;
    for (int d = 0; d < H; ++d) sp += hs[tid][d] * wc[d] * u[d];
    pv[tid] = sp;
  }
  if (tid == 0) {
    float S1 = 0.0f, S2 = 0.0f;
    for (int j = 0; j < L; ++j) { S1 += Wmy_w[j]; S2 += Wmy_w[j] * bv[j]; }
    red[0] = S1; red[1] = S2;
  }
  __syncthreads();
  if (tid < L) {
    float S1 = red[0], S2 = red[1], cb = wCo_b[0];
    smv[tid] = av[tid] * S1 + S2 + pv[tid] + cb * S1
             - Wmy_w[tid] * (av[tid] + bv[tid] + qv[tid] + cb) + Wmy_b[0];
  }
  __syncthreads();
  if (tid == 0) {                              // softmax over i (75 values)
    float mx = smv[0];
    for (int i = 1; i < L; ++i) mx = fmaxf(mx, smv[i]);
    float s = 0.0f;
    for (int i = 0; i < L; ++i) { attn[i] = __expf(smv[i] - mx); s += attn[i]; }
    float inv = 1.0f / s;
    for (int i = 0; i < L; ++i) attn[i] *= inv;
  }
  __syncthreads();
  for (int d = tid; d < H; d += 256) {         // pooled = attn @ h
    float s = 0.0f;
    for (int i = 0; i < L; ++i) s += attn[i] * hs[i][d];
    pooled[d] = s;
  }
  __syncthreads();
  if (tid < OUT) {                             // logits
    float s = logits_b[tid];
    for (int d = 0; d < H; ++d) s += pooled[d] * logits_w[tid * H + d];
    out[b * OUT + tid] = s;
  }
}

// ---------------------------------------------------------------------------
extern "C" void kernel_launch(void* const* d_in, const int* in_sizes, int n_in,
                              void* d_out, int out_size, void* d_ws, size_t ws_size,
                              hipStream_t stream) {
  const int*   seq      = (const int*)  d_in[0];
  const float* emb      = (const float*)d_in[1];
  const float* W_ih     = (const float*)d_in[2];
  const float* W_hh     = (const float*)d_in[3];
  const float* b_ih     = (const float*)d_in[4];
  const float* b_hh     = (const float*)d_in[5];
  const float* wCo_w    = (const float*)d_in[6];
  const float* wCo_b    = (const float*)d_in[7];
  const float* Wmy_w    = (const float*)d_in[8];
  const float* Wmy_b    = (const float*)d_in[9];
  const float* logits_w = (const float*)d_in[10];
  const float* logits_b = (const float*)d_in[11];
  float* out = (float*)d_out;

  char* ws = (char*)d_ws;
  float* h_all = (float*)ws;
  size_t off = (size_t)BATCH * L * H * sizeof(float);
  off = (off + 255) & ~(size_t)255;
  unsigned short* WihP = (unsigned short*)(ws + off);
  off += (size_t)3 * NP * KP * sizeof(unsigned short);
  off = (off + 255) & ~(size_t)255;
  unsigned short* WhhP = (unsigned short*)(ws + off);

  const int nprep = 3 * NP * KP;
  prep_w_kernel<<<(nprep + 255) / 256, 256, 0, stream>>>(W_ih, WihP);
  prep_w_kernel<<<(nprep + 255) / 256, 256, 0, stream>>>(W_hh, WhhP);
  gru_kernel<<<BATCH / BT, 256, 0, stream>>>(seq, emb, WihP, WhhP,
                                             b_ih, b_hh, h_all);
  coattn_kernel<<<BATCH, 256, 0, stream>>>(h_all, wCo_w, wCo_b, Wmy_w, Wmy_b,
                                           logits_w, logits_b, out);
}
